// KensertGCN_54597624267394
// MI455X (gfx1250) — compile-verified, matrix operand order checked by reading, not run on hardware
//
#include <hip/hip_runtime.h>
#include <hip/hip_bf16.h>

#define N_NODES 100000
#define N_EDGES 3200000
#define N_GRAPHS 2048
#define BN_EPS 1e-5f

typedef __attribute__((ext_vector_type(16))) __bf16 v16bf;
typedef __attribute__((ext_vector_type(8)))  __bf16 v8bf;
typedef __attribute__((ext_vector_type(4)))  __bf16 v4bf;
typedef __attribute__((ext_vector_type(8)))  float  v8f;

__device__ __forceinline__ void atomAddF(float* p, float v) {
    unsafeAtomicAdd(p, v);                            // global_atomic_add_f32
}

// -------------------------------------------------------------------------
// f32 -> packed bf16 (4 elems/thread, b128 in / b64 out)
// -------------------------------------------------------------------------
__global__ void cvt_bf16_kernel(const float* __restrict__ src, __bf16* __restrict__ dst,
                                size_t total) {
    size_t i = ((size_t)blockIdx.x * blockDim.x + threadIdx.x) * 4;
    if (i >= total) return;
    float4 v = *(const float4*)(src + i);
    v4bf o = { (__bf16)v.x, (__bf16)v.y, (__bf16)v.z, (__bf16)v.w };
    *(v4bf*)(dst + i) = o;
}

// -------------------------------------------------------------------------
// Weight prep: W[K,N] f32 -> WT[N,K] bf16 (reads coalesced along N)
// -------------------------------------------------------------------------
__global__ void tconv_kernel(const float* __restrict__ W, __bf16* __restrict__ WT,
                             int K, int N) {
    int idx = blockIdx.x * blockDim.x + threadIdx.x;
    if (idx >= K * N) return;
    int k = idx / N, n = idx % N;
    WT[(size_t)n * K + k] = (__bf16)W[idx];
}

// -------------------------------------------------------------------------
// SpMM: agg[row] += val * h[col]   (h in bf16, v8bf gathers, fp32 atomics)
// -------------------------------------------------------------------------
__global__ void spmm_kernel(const int* __restrict__ row, const int* __restrict__ col,
                            const float* __restrict__ val, const __bf16* __restrict__ h,
                            float* __restrict__ agg, int C) {
    const int cpt = C >> 3;                           // 8-elem chunks per edge
    size_t idx = (size_t)blockIdx.x * blockDim.x + threadIdx.x;
    size_t total = (size_t)N_EDGES * cpt;
    if (idx >= total) return;
    int e = (int)(idx / cpt);
    int c = (int)(idx % cpt) * 8;
    float v = val[e];
    int r = row[e], s = col[e];
    v8bf hv = *(const v8bf*)(h + (size_t)s * C + c);
    float* dst = agg + (size_t)r * C + c;
#pragma unroll
    for (int j = 0; j < 8; ++j)
        atomAddF(dst + j, v * (float)hv[j]);
}

// -------------------------------------------------------------------------
// GEMM: C[M,N] = A[M,K] @ B[K,N] + bias (opt relu), A bf16, BT = B^T bf16.
// One wave computes a 16x64 strip: 4 WMMA accumulators sharing the A frag.
// M%16==0, N%64==0, K%32==0. Inner loop: 10 b128 loads + 4 v_wmma.
// -------------------------------------------------------------------------
__global__ void gemm_wmma_kernel(const __bf16* __restrict__ A, const __bf16* __restrict__ BT,
                                 const float* __restrict__ bias, float* __restrict__ C,
                                 int M, int K, int N, int relu) {
    const int g4 = N >> 6;                            // 64-column groups
    int waveId = blockIdx.x * (blockDim.x >> 5) + (threadIdx.x >> 5);
    if (waveId >= (M >> 4) * g4) return;              // wave-uniform exit
    const int tileM = waveId / g4;
    const int grpN  = waveId % g4;

    const int lane = threadIdx.x & 31;
    const int lm   = lane & 15;
    const int half = lane >> 4;

    const __bf16* arow = A + (size_t)(tileM * 16 + lm) * K;
    const __bf16* b0 = BT + (size_t)(grpN * 64 + 0  + lm) * K;
    const __bf16* b1 = BT + (size_t)(grpN * 64 + 16 + lm) * K;
    const __bf16* b2 = BT + (size_t)(grpN * 64 + 32 + lm) * K;
    const __bf16* b3 = BT + (size_t)(grpN * 64 + 48 + lm) * K;

    v8f acc0 = {}, acc1 = {}, acc2 = {}, acc3 = {};
    for (int k0 = 0; k0 < K; k0 += 32) {
        const int kg0 = k0 + half * 8;
        const int kg1 = kg0 + 16;
        v8bf aL = *(const v8bf*)(arow + kg0);
        v8bf aH = *(const v8bf*)(arow + kg1);
        v16bf av = __builtin_shufflevector(aL, aH, 0,1,2,3,4,5,6,7,8,9,10,11,12,13,14,15);

        v8bf tL, tH; v16bf bv;
        tL = *(const v8bf*)(b0 + kg0); tH = *(const v8bf*)(b0 + kg1);
        bv = __builtin_shufflevector(tL, tH, 0,1,2,3,4,5,6,7,8,9,10,11,12,13,14,15);
        acc0 = __builtin_amdgcn_wmma_f32_16x16x32_bf16(false, av, false, bv, (short)0, acc0, false, false);

        tL = *(const v8bf*)(b1 + kg0); tH = *(const v8bf*)(b1 + kg1);
        bv = __builtin_shufflevector(tL, tH, 0,1,2,3,4,5,6,7,8,9,10,11,12,13,14,15);
        acc1 = __builtin_amdgcn_wmma_f32_16x16x32_bf16(false, av, false, bv, (short)0, acc1, false, false);

        tL = *(const v8bf*)(b2 + kg0); tH = *(const v8bf*)(b2 + kg1);
        bv = __builtin_shufflevector(tL, tH, 0,1,2,3,4,5,6,7,8,9,10,11,12,13,14,15);
        acc2 = __builtin_amdgcn_wmma_f32_16x16x32_bf16(false, av, false, bv, (short)0, acc2, false, false);

        tL = *(const v8bf*)(b3 + kg0); tH = *(const v8bf*)(b3 + kg1);
        bv = __builtin_shufflevector(tL, tH, 0,1,2,3,4,5,6,7,8,9,10,11,12,13,14,15);
        acc3 = __builtin_amdgcn_wmma_f32_16x16x32_bf16(false, av, false, bv, (short)0, acc3, false, false);
    }

    // D layout: elem j -> row tileM*16 + half*8 + j, col = lane&15 within tile
    const int rbase = tileM * 16 + half * 8;
    v8f accs[4] = {acc0, acc1, acc2, acc3};
#pragma unroll
    for (int t = 0; t < 4; ++t) {
        const int col = grpN * 64 + t * 16 + lm;
        const float bb = bias ? bias[col] : 0.0f;
#pragma unroll
        for (int j = 0; j < 8; ++j) {
            float v = accs[t][j] + bb;
            if (relu) v = fmaxf(v, 0.0f);
            C[(size_t)(rbase + j) * N + col] = v;
        }
    }
}

// -------------------------------------------------------------------------
// BatchNorm stats: per-column sum & sumsq (coalesced, LDS-staged atomics)
// -------------------------------------------------------------------------
__global__ void bn_stats_kernel(const float* __restrict__ x, float* __restrict__ sums,
                                float* __restrict__ sqs, size_t total, int C) {
    __shared__ float ssum[512];
    __shared__ float ssq[512];
    for (int i = threadIdx.x; i < C; i += blockDim.x) { ssum[i] = 0.f; ssq[i] = 0.f; }
    __syncthreads();
    for (size_t i = (size_t)blockIdx.x * blockDim.x + threadIdx.x; i < total;
         i += (size_t)gridDim.x * blockDim.x) {
        float v = x[i];
        int c = (int)(i % C);
        atomicAdd(&ssum[c], v);
        atomicAdd(&ssq[c], v * v);
    }
    __syncthreads();
    for (int i = threadIdx.x; i < C; i += blockDim.x) {
        if (ssum[i] != 0.f) atomAddF(&sums[i], ssum[i]);
        if (ssq[i]  != 0.f) atomAddF(&sqs[i],  ssq[i]);
    }
}

__global__ void bn_finalize_kernel(const float* __restrict__ sums, const float* __restrict__ sqs,
                                   const float* __restrict__ gamma, const float* __restrict__ beta,
                                   float* __restrict__ scale, float* __restrict__ shift,
                                   int C, float invN) {
    int c = blockIdx.x * blockDim.x + threadIdx.x;
    if (c >= C) return;
    float mu  = sums[c] * invN;
    float var = sqs[c] * invN - mu * mu;
    float s   = gamma[c] * rsqrtf(var + BN_EPS);
    scale[c] = s;
    shift[c] = beta[c] - mu * s;
}

// BN affine + ReLU, emit bf16 h for the next SpMM / pool (4 elems/thread)
__global__ void bn_relu_bf16_kernel(const float* __restrict__ x, const float* __restrict__ scale,
                                    const float* __restrict__ shift, __bf16* __restrict__ y,
                                    size_t total, int C) {
    size_t i = ((size_t)blockIdx.x * blockDim.x + threadIdx.x) * 4;
    if (i >= total) return;
    int c = (int)(i % C);                             // C%4==0 so chunk stays in-row
    float4 v = *(const float4*)(x + i);
    float4 sc = *(const float4*)(scale + c);
    float4 sh = *(const float4*)(shift + c);
    v4bf o = { (__bf16)fmaxf(v.x * sc.x + sh.x, 0.f),
               (__bf16)fmaxf(v.y * sc.y + sh.y, 0.f),
               (__bf16)fmaxf(v.z * sc.z + sh.z, 0.f),
               (__bf16)fmaxf(v.w * sc.w + sh.w, 0.f) };
    *(v4bf*)(y + i) = o;
}

// -------------------------------------------------------------------------
// global_add_pool: fp[batch[n]] += h[n]   (h bf16, fp32 atomics, C = 512)
// -------------------------------------------------------------------------
__global__ void pool_kernel(const __bf16* __restrict__ h, const int* __restrict__ batch,
                            float* __restrict__ fp) {
    const int C = 512, cpt = 64;                      // 8-elem chunks
    size_t idx = (size_t)blockIdx.x * blockDim.x + threadIdx.x;
    size_t total = (size_t)N_NODES * cpt;
    if (idx >= total) return;
    int n = (int)(idx / cpt);
    int c = (int)(idx % cpt) * 8;
    int g = batch[n];
    v8bf hv = *(const v8bf*)(h + (size_t)n * C + c);
    float* dst = fp + (size_t)g * C + c;
#pragma unroll
    for (int j = 0; j < 8; ++j)
        atomAddF(dst + j, (float)hv[j]);
}

// -------------------------------------------------------------------------
// Final layer: y[g] = dot(x[g,:K], w) + b   (block per graph, fp32)
// -------------------------------------------------------------------------
__global__ void final_dot_kernel(const float* __restrict__ x, const float* __restrict__ w,
                                 const float* __restrict__ b, float* __restrict__ y, int K) {
    __shared__ float red[128];
    int g = blockIdx.x;
    float s = 0.f;
    for (int k = threadIdx.x; k < K; k += blockDim.x)
        s += x[(size_t)g * K + k] * w[k];
    red[threadIdx.x] = s;
    __syncthreads();
    for (int off = 64; off > 0; off >>= 1) {
        if ((int)threadIdx.x < off) red[threadIdx.x] += red[threadIdx.x + off];
        __syncthreads();
    }
    if (threadIdx.x == 0) y[g] = red[0] + b[0];
}

// -------------------------------------------------------------------------
extern "C" void kernel_launch(void* const* d_in, const int* in_sizes, int n_in,
                              void* d_out, int out_size, void* d_ws, size_t ws_size,
                              hipStream_t stream) {
    const float* node_attr = (const float*)d_in[0];
    const int*   adj       = (const int*)d_in[1];
    const float* adj_value = (const float*)d_in[2];
    const int*   batch     = (const int*)d_in[3];
    const float *Wg[3], *bg[3], *gamma[3], *beta[3];
    for (int i = 0; i < 3; ++i) {
        Wg[i]    = (const float*)d_in[4 + 4 * i];
        bg[i]    = (const float*)d_in[5 + 4 * i];
        gamma[i] = (const float*)d_in[6 + 4 * i];
        beta[i]  = (const float*)d_in[7 + 4 * i];
    }
    const float* Wp0 = (const float*)d_in[16];
    const float* bp0 = (const float*)d_in[17];
    const float* Wp1 = (const float*)d_in[18];
    const float* bp1 = (const float*)d_in[19];
    const float* Wp2 = (const float*)d_in[20];
    const float* bp2 = (const float*)d_in[21];
    float* out = (float*)d_out;

    const int* rowp = adj;
    const int* colp = adj + N_EDGES;

    // ---- workspace carve-up (256B aligned) ----
    char* ws = (char*)d_ws;
    size_t off = 0;
    auto carveF = [&](size_t n) -> float* {
        float* p = (float*)(ws + off); off += ((n * 4 + 255) / 256) * 256; return p;
    };
    auto carveH = [&](size_t n) -> __bf16* {
        __bf16* p = (__bf16*)(ws + off); off += ((n * 2 + 255) / 256) * 256; return p;
    };
    float*  linF  = carveF((size_t)N_NODES * 512);    // SpMM accum / GEMM out (fp32)
    __bf16* aggH  = carveH((size_t)N_NODES * 512);    // bf16 copy of SpMM result
    __bf16* hB    = carveH((size_t)N_NODES * 512);    // bf16 activations
    float*  fp    = carveF((size_t)N_GRAPHS * 512);
    __bf16* fpH   = carveH((size_t)N_GRAPHS * 512);
    float*  m1    = carveF((size_t)N_GRAPHS * 1024);
    __bf16* m1H   = carveH((size_t)N_GRAPHS * 1024);
    float*  m2    = carveF((size_t)N_GRAPHS * 512);
    float*  sums  = carveF(512);
    float*  sqs   = carveF(512);
    float*  scale = carveF(512);
    float*  shift = carveF(512);
    __bf16* wgT[3];
    wgT[0] = carveH(128 * 256);
    wgT[1] = carveH(256 * 512);
    wgT[2] = carveH(512 * 512);
    __bf16* wpT0 = carveH(512 * 1024);
    __bf16* wpT1 = carveH(1024 * 512);
    (void)ws_size; (void)in_sizes; (void)n_in; (void)out_size;

    const int gcn_in[3]  = {128, 256, 512};
    const int gcn_out[3] = {256, 512, 512};
    const float invN = 1.0f / (float)N_NODES;

    // ---- weight prep: f32 [K,N] -> bf16 [N,K] ----
    for (int i = 0; i < 3; ++i) {
        int kn = gcn_in[i] * gcn_out[i];
        tconv_kernel<<<(kn + 255) / 256, 256, 0, stream>>>(Wg[i], wgT[i], gcn_in[i], gcn_out[i]);
    }
    tconv_kernel<<<(512 * 1024 + 255) / 256, 256, 0, stream>>>(Wp0, wpT0, 512, 1024);
    tconv_kernel<<<(1024 * 512 + 255) / 256, 256, 0, stream>>>(Wp1, wpT1, 1024, 512);

    // ---- input features -> bf16 ----
    {
        size_t total = (size_t)N_NODES * 128;
        cvt_bf16_kernel<<<(unsigned)((total / 4 + 255) / 256), 256, 0, stream>>>(
            node_attr, hB, total);
    }

    // ---- GCN layers ----
    for (int L = 0; L < 3; ++L) {
        const int Cin = gcn_in[L], Cout = gcn_out[L];
        size_t aggElems = (size_t)N_NODES * Cin;
        // 1) SpMM (bf16 gather -> fp32 atomic scatter)
        hipMemsetAsync(linF, 0, aggElems * 4, stream);
        {
            size_t total = (size_t)N_EDGES * (Cin >> 3);
            spmm_kernel<<<(unsigned)((total + 255) / 256), 256, 0, stream>>>(
                rowp, colp, adj_value, hB, linF, Cin);
        }
        // 2) agg -> bf16
        cvt_bf16_kernel<<<(unsigned)((aggElems / 4 + 255) / 256), 256, 0, stream>>>(
            linF, aggH, aggElems);
        // 3) GEMM: linF = aggH @ Wg + bg   (fp32 out; BN follows)
        {
            unsigned waves = (unsigned)((N_NODES / 16) * (Cout / 64));
            gemm_wmma_kernel<<<(waves + 7) / 8, 256, 0, stream>>>(
                aggH, wgT[L], bg[L], linF, N_NODES, Cin, Cout, 0);
        }
        // 4) BN stats + fused affine/ReLU -> bf16 h
        hipMemsetAsync(sums, 0, Cout * 4, stream);
        hipMemsetAsync(sqs, 0, Cout * 4, stream);
        {
            size_t total = (size_t)N_NODES * Cout;
            bn_stats_kernel<<<1024, 256, 0, stream>>>(linF, sums, sqs, total, Cout);
            bn_finalize_kernel<<<(Cout + 255) / 256, 256, 0, stream>>>(
                sums, sqs, gamma[L], beta[L], scale, shift, Cout, invN);
            bn_relu_bf16_kernel<<<(unsigned)((total / 4 + 255) / 256), 256, 0, stream>>>(
                linF, scale, shift, hB, total, Cout);
        }
    }

    // ---- global_add_pool -> fp [2048, 512] ----
    hipMemsetAsync(fp, 0, (size_t)N_GRAPHS * 512 * 4, stream);
    {
        size_t total = (size_t)N_NODES * 64;
        pool_kernel<<<(unsigned)((total + 255) / 256), 256, 0, stream>>>(hB, batch, fp);
    }

    // ---- predictor MLP ----
    {
        size_t t0 = (size_t)N_GRAPHS * 512;
        cvt_bf16_kernel<<<(unsigned)((t0 / 4 + 255) / 256), 256, 0, stream>>>(fp, fpH, t0);
        unsigned waves0 = (unsigned)((N_GRAPHS / 16) * (1024 / 64));
        gemm_wmma_kernel<<<(waves0 + 7) / 8, 256, 0, stream>>>(
            fpH, wpT0, bp0, m1, N_GRAPHS, 512, 1024, 1);

        size_t t1 = (size_t)N_GRAPHS * 1024;
        cvt_bf16_kernel<<<(unsigned)((t1 / 4 + 255) / 256), 256, 0, stream>>>(m1, m1H, t1);
        unsigned waves1 = (unsigned)((N_GRAPHS / 16) * (512 / 64));
        gemm_wmma_kernel<<<(waves1 + 7) / 8, 256, 0, stream>>>(
            m1H, wpT1, bp1, m2, N_GRAPHS, 1024, 512, 1);

        final_dot_kernel<<<N_GRAPHS, 128, 0, stream>>>(m2, Wp2, bp2, out, 512);
    }
}
